// TGATModel_91139206021476
// MI455X (gfx1250) — matrix operand order repaired and checked by reading, not compiled, not run on hardware
//
#include <hip/hip_runtime.h>

// Problem constants (from reference)
#define NN 20000
#define EE 320000
#define TT 8
#define HH 32
#define ETOT (EE + NN)          // edges + self loops
#define ENC_NEG_INF 0x007FFFFFu // order-preserving encoding of -inf

typedef __attribute__((ext_vector_type(16))) _Float16 v16h;
typedef __attribute__((ext_vector_type(8)))  _Float16 v8h;
typedef __attribute__((ext_vector_type(8)))  float    v8f;

// ---- helpers -------------------------------------------------------------
__device__ __forceinline__ unsigned enc_f(float f) {
    unsigned u = __float_as_uint(f);
    return (u & 0x80000000u) ? ~u : (u | 0x80000000u);
}
__device__ __forceinline__ float dec_f(unsigned u) {
    return (u & 0x80000000u) ? __uint_as_float(u & 0x7FFFFFFFu)
                             : __uint_as_float(~u);
}
__device__ __forceinline__ float lrelu(float v) { return v > 0.f ? v : 0.2f * v; }
__device__ __forceinline__ float eluf(float v)  { return v > 0.f ? v : (__expf(v) - 1.f); }
__device__ __forceinline__ float sigm(float v)  { return 1.f / (1.f + __expf(-v)); }

__device__ __forceinline__ v16h cat8(v8h a, v8h b) {
    v16h r;
#pragma unroll
    for (int i = 0; i < 8; i++) { r[i] = a[i]; r[8 + i] = b[i]; }
    return r;
}

__device__ __forceinline__ int edge_src(const int* ei, int t, int e) {
    return (e < EE) ? ei[(size_t)t * 2 * EE + e] : (e - EE);
}
__device__ __forceinline__ int edge_dst(const int* ei, int t, int e) {
    return (e < EE) ? ei[(size_t)t * 2 * EE + EE + e] : (e - EE);
}

// ---- init ---------------------------------------------------------------
__global__ void k_zero(float* p, int n) {
    int i = blockIdx.x * blockDim.x + threadIdx.x;
    if (i < n) p[i] = 0.f;
}

// ---- weight prep: f16 copies in WMMA-friendly layouts (once per launch) --
__global__ void k_prep_w(const float* W2, const float* wih, const float* whh,
                         _Float16* W2T, _Float16* wihh, _Float16* whhh) {
    int i = blockIdx.x * blockDim.x + threadIdx.x;
    if (i < 64 * 32) {               // W2T[col*64 + k] = W2[k*32 + col]
        int col = i >> 6, k = i & 63;
        W2T[i] = (_Float16)W2[k * 32 + col];
    }
    if (i < 96 * 32) {               // torch layout already K-contiguous
        wihh[i] = (_Float16)wih[i];
        whhh[i] = (_Float16)whh[i];
    }
}

// ---- GAT layer 1: node prep (h1 = x*W1, attn coeffs, reset accumulators) -
__global__ void k_gat1_node(const float* x, const float* W1,
                            const float* as1, const float* ad1,
                            float* h1, float* a1s, float* a1d,
                            unsigned* m1, float* den1, float* agg1, int t) {
    int n = blockIdx.x * blockDim.x + threadIdx.x;
    if (n >= NN) return;
    float xv = x[(size_t)t * NN + n];
    float s0 = 0.f, s1 = 0.f, d0 = 0.f, d1 = 0.f;
#pragma unroll
    for (int j = 0; j < 64; j++) {
        float hv = xv * W1[j];
        h1[(size_t)n * 64 + j] = hv;
        agg1[(size_t)n * 64 + j] = 0.f;
        if (j < 32) { s0 += hv * as1[j]; d0 += hv * ad1[j]; }
        else        { s1 += hv * as1[j]; d1 += hv * ad1[j]; }
    }
    a1s[n * 2 + 0] = s0; a1s[n * 2 + 1] = s1;
    a1d[n * 2 + 0] = d0; a1d[n * 2 + 1] = d1;
    m1[n * 2 + 0] = ENC_NEG_INF; m1[n * 2 + 1] = ENC_NEG_INF;
    den1[n * 2 + 0] = 0.f; den1[n * 2 + 1] = 0.f;
}

// ---- GAT edge passes (2-head version, layer 1) --------------------------
__global__ void k_e1_max(const int* ei, const float* a1s, const float* a1d,
                         unsigned* m1, int t) {
    int e = blockIdx.x * blockDim.x + threadIdx.x;
    if (e >= ETOT) return;
    int s = edge_src(ei, t, e), d = edge_dst(ei, t, e);
#pragma unroll
    for (int h = 0; h < 2; h++) {
        float ev = lrelu(a1s[s * 2 + h] + a1d[d * 2 + h]);
        atomicMax(&m1[d * 2 + h], enc_f(ev));
    }
}
__global__ void k_e1_sum(const int* ei, const float* a1s, const float* a1d,
                         const unsigned* m1, float* den1, int t) {
    int e = blockIdx.x * blockDim.x + threadIdx.x;
    if (e >= ETOT) return;
    int s = edge_src(ei, t, e), d = edge_dst(ei, t, e);
#pragma unroll
    for (int h = 0; h < 2; h++) {
        float ev = lrelu(a1s[s * 2 + h] + a1d[d * 2 + h]);
        atomicAdd(&den1[d * 2 + h], __expf(ev - dec_f(m1[d * 2 + h])));
    }
}
__global__ void k_e1_msg(const int* ei, const float* a1s, const float* a1d,
                         const unsigned* m1, const float* den1,
                         const float* h1, float* agg1, int t) {
    int tid = blockIdx.x * blockDim.x + threadIdx.x;
    int e = tid >> 6, j = tid & 63;
    if (e >= ETOT) return;
    int s = edge_src(ei, t, e), d = edge_dst(ei, t, e);
    int h = j >> 5;
    float ev = lrelu(a1s[s * 2 + h] + a1d[d * 2 + h]);
    float ex = __expf(ev - dec_f(m1[d * 2 + h]));
    float alpha = ex / (den1[d * 2 + h] + 1e-16f);
    atomicAdd(&agg1[(size_t)d * 64 + j], h1[(size_t)s * 64 + j] * alpha);
}

__global__ void k_fin1(const float* agg1, const float* b1, _Float16* x2h) {
    int i = blockIdx.x * blockDim.x + threadIdx.x;
    if (i < NN * 64) x2h[i] = (_Float16)eluf(agg1[i] + b1[i & 63]);
}

// ---- GAT2 linear: [N,64] @ [64,32] via WMMA f16, wave per 16-row tile ---
// A operand: x2h rows, lane-run layout is 8-element contiguous K runs.
// B operand: W2T[col*64 + k] — two 8-element K runs per lane (same order as A).
__global__ void k_gat2_gemm(const _Float16* x2h, const _Float16* W2T, float* h2) {
    int tile = blockIdx.x;           // 1250 tiles
    int lane = threadIdx.x;          // 0..31
    int M = lane & 15, hi = lane >> 4;
    int row0 = tile * 16;
    const _Float16* arow = x2h + (size_t)(row0 + M) * 64;
    v16h A[2];
#pragma unroll
    for (int q = 0; q < 2; q++) {
        v8h lo = *(const v8h*)(arow + q * 32 + hi * 8);        // K = q*32+hi*8 ..+7
        v8h up = *(const v8h*)(arow + q * 32 + 16 + hi * 8);   // K = q*32+16+hi*8 ..+7
        A[q] = cat8(lo, up);
    }
#pragma unroll
    for (int c = 0; c < 2; c++) {
        int col = c * 16 + M;
        v8f acc = {};
#pragma unroll
        for (int q = 0; q < 2; q++) {
            const _Float16* bp = W2T + (size_t)col * 64 + q * 32 + hi * 8;
            v16h B = cat8(*(const v8h*)bp, *(const v8h*)(bp + 16));
            acc = __builtin_amdgcn_wmma_f32_16x16x32_f16(
                false, A[q], false, B, (short)0, acc, false, false);
        }
#pragma unroll
        for (int r = 0; r < 8; r++)
            h2[(size_t)(row0 + r + hi * 8) * 32 + col] = acc[r];
    }
}

// ---- GAT layer 2: node prep (1 head) ------------------------------------
__global__ void k_gat2_node(const float* h2, const float* as2, const float* ad2,
                            float* a2s, float* a2d, unsigned* m2, float* den2,
                            float* agg2) {
    int n = blockIdx.x * blockDim.x + threadIdx.x;
    if (n >= NN) return;
    float s = 0.f, d = 0.f;
#pragma unroll
    for (int j = 0; j < 32; j++) {
        float hv = h2[(size_t)n * 32 + j];
        s += hv * as2[j]; d += hv * ad2[j];
        agg2[(size_t)n * 32 + j] = 0.f;
    }
    a2s[n] = s; a2d[n] = d;
    m2[n] = ENC_NEG_INF; den2[n] = 0.f;
}
__global__ void k_e2_max(const int* ei, const float* a2s, const float* a2d,
                         unsigned* m2, int t) {
    int e = blockIdx.x * blockDim.x + threadIdx.x;
    if (e >= ETOT) return;
    int s = edge_src(ei, t, e), d = edge_dst(ei, t, e);
    atomicMax(&m2[d], enc_f(lrelu(a2s[s] + a2d[d])));
}
__global__ void k_e2_sum(const int* ei, const float* a2s, const float* a2d,
                         const unsigned* m2, float* den2, int t) {
    int e = blockIdx.x * blockDim.x + threadIdx.x;
    if (e >= ETOT) return;
    int s = edge_src(ei, t, e), d = edge_dst(ei, t, e);
    atomicAdd(&den2[d], __expf(lrelu(a2s[s] + a2d[d]) - dec_f(m2[d])));
}
__global__ void k_e2_msg(const int* ei, const float* a2s, const float* a2d,
                         const unsigned* m2, const float* den2,
                         const float* h2, float* agg2, int t) {
    int tid = blockIdx.x * blockDim.x + threadIdx.x;
    int e = tid >> 5, j = tid & 31;
    if (e >= ETOT) return;
    int s = edge_src(ei, t, e), d = edge_dst(ei, t, e);
    float ex = __expf(lrelu(a2s[s] + a2d[d]) - dec_f(m2[d]));
    float alpha = ex / (den2[d] + 1e-16f);
    atomicAdd(&agg2[(size_t)d * 32 + j], h2[(size_t)s * 32 + j] * alpha);
}
__global__ void k_fin2(const float* agg2, const float* b2, _Float16* ench) {
    int i = blockIdx.x * blockDim.x + threadIdx.x;
    if (i < NN * 32) ench[i] = (_Float16)eluf(agg2[i] + b2[i & 31]);
}

// ---- fused GRU step: 12 WMMAs + gates + h update, wave per 16 nodes -----
__global__ void k_gru(const _Float16* ench, float* hst, _Float16* hsth,
                      const _Float16* wihh, const _Float16* whhh,
                      const float* bih, const float* bhh) {
    int tile = blockIdx.x;           // 1250
    int lane = threadIdx.x;          // 0..31
    int M = lane & 15, hi = lane >> 4;
    int row0 = tile * 16;
    const _Float16* xrow = ench + (size_t)(row0 + M) * 32;
    const _Float16* hrow = hsth + (size_t)(row0 + M) * 32;
    v16h Ax = cat8(*(const v8h*)(xrow + hi * 8), *(const v8h*)(xrow + 16 + hi * 8));
    v16h Ah = cat8(*(const v8h*)(hrow + hi * 8), *(const v8h*)(hrow + 16 + hi * 8));
    v8f gi[6], gh[6];
#pragma unroll
    for (int ct = 0; ct < 6; ct++) {
        int col = ct * 16 + M;       // gate row index 0..95 (r,z,n)
        float bi = bih[col], bh = bhh[col];
        v8f ci, ch;
#pragma unroll
        for (int r = 0; r < 8; r++) { ci[r] = bi; ch[r] = bh; }
        const _Float16* bip = wihh + (size_t)col * 32 + hi * 8;
        const _Float16* bhp = whhh + (size_t)col * 32 + hi * 8;
        v16h Bi = cat8(*(const v8h*)bip, *(const v8h*)(bip + 16));
        v16h Bh = cat8(*(const v8h*)bhp, *(const v8h*)(bhp + 16));
        gi[ct] = __builtin_amdgcn_wmma_f32_16x16x32_f16(
            false, Ax, false, Bi, (short)0, ci, false, false);
        gh[ct] = __builtin_amdgcn_wmma_f32_16x16x32_f16(
            false, Ah, false, Bh, (short)0, ch, false, false);
    }
#pragma unroll
    for (int p = 0; p < 2; p++) {
#pragma unroll
        for (int r = 0; r < 8; r++) {
            int m = row0 + r + hi * 8;
            int j = p * 16 + M;
            float rg = sigm(gi[p][r]     + gh[p][r]);
            float zg = sigm(gi[2 + p][r] + gh[2 + p][r]);
            float ng = tanhf(gi[4 + p][r] + rg * gh[4 + p][r]);
            float hv = hst[(size_t)m * 32 + j];
            float hnew = (1.f - zg) * ng + zg * hv;
            hst[(size_t)m * 32 + j]  = hnew;
            hsth[(size_t)m * 32 + j] = (_Float16)hnew;
        }
    }
}

// ---- output head --------------------------------------------------------
__global__ void k_fc(const float* hst, const float* fw, const float* fb,
                     float* out) {
    int n = blockIdx.x * blockDim.x + threadIdx.x;
    if (n >= NN) return;
    float s = fb[0];
#pragma unroll
    for (int j = 0; j < 32; j++) s += hst[(size_t)n * 32 + j] * fw[j];
    out[n] = s;
}

// ---- host-side orchestration --------------------------------------------
extern "C" void kernel_launch(void* const* d_in, const int* in_sizes, int n_in,
                              void* d_out, int out_size, void* d_ws, size_t ws_size,
                              hipStream_t stream) {
    (void)in_sizes; (void)n_in; (void)out_size; (void)ws_size;
    const float* x   = (const float*)d_in[0];
    const int*   ei  = (const int*)  d_in[1];
    const float* W1  = (const float*)d_in[2];
    const float* as1 = (const float*)d_in[3];
    const float* ad1 = (const float*)d_in[4];
    const float* b1  = (const float*)d_in[5];
    const float* W2  = (const float*)d_in[6];
    const float* as2 = (const float*)d_in[7];
    const float* ad2 = (const float*)d_in[8];
    const float* b2  = (const float*)d_in[9];
    const float* wih = (const float*)d_in[10];
    const float* whh = (const float*)d_in[11];
    const float* bih = (const float*)d_in[12];
    const float* bhh = (const float*)d_in[13];
    const float* fw  = (const float*)d_in[14];
    const float* fb  = (const float*)d_in[15];
    float* out = (float*)d_out;

    // workspace carve-up (floats); ~24 MB total. All sub-buffers 16B aligned.
    float* ws = (float*)d_ws;
    size_t o = 0;
    float*     h1   = ws + o; o += (size_t)NN * 64;
    float*     a1s  = ws + o; o += (size_t)NN * 2;
    float*     a1d  = ws + o; o += (size_t)NN * 2;
    unsigned*  m1   = (unsigned*)(ws + o); o += (size_t)NN * 2;
    float*     den1 = ws + o; o += (size_t)NN * 2;
    float*     agg1 = ws + o; o += (size_t)NN * 64;
    _Float16*  x2h  = (_Float16*)(ws + o); o += (size_t)NN * 32;  // N*64 f16
    float*     h2   = ws + o; o += (size_t)NN * 32;
    float*     a2s  = ws + o; o += (size_t)NN;
    float*     a2d  = ws + o; o += (size_t)NN;
    unsigned*  m2   = (unsigned*)(ws + o); o += (size_t)NN;
    float*     den2 = ws + o; o += (size_t)NN;
    float*     agg2 = ws + o; o += (size_t)NN * 32;
    _Float16*  ench = (_Float16*)(ws + o); o += (size_t)NN * 16;  // N*32 f16
    float*     hst  = ws + o; o += (size_t)NN * 32;
    _Float16*  hsth = (_Float16*)(ws + o); o += (size_t)NN * 16;  // N*32 f16
    _Float16*  W2T  = (_Float16*)(ws + o); o += 1024;             // 32x64 f16
    _Float16*  wihh = (_Float16*)(ws + o); o += 1536;             // 96x32 f16
    _Float16*  whhh = (_Float16*)(ws + o); o += 1536;             // 96x32 f16

    const int B = 256;
    const int gN    = (NN + B - 1) / B;
    const int gE    = (ETOT + B - 1) / B;
    const int gE64  = (ETOT * 64 + B - 1) / B;
    const int gE32  = (ETOT * 32 + B - 1) / B;
    const int gN64  = (NN * 64 + B - 1) / B;
    const int gN32  = (NN * 32 + B - 1) / B;
    const int gTile = NN / 16;   // 1250, exact

    k_zero<<<gN32, B, 0, stream>>>(hst, NN * 32);
    k_zero<<<(NN * 16 + B - 1) / B, B, 0, stream>>>((float*)hsth, NN * 16);
    k_prep_w<<<(96 * 32 + B - 1) / B, B, 0, stream>>>(W2, wih, whh, W2T, wihh, whhh);

    for (int t = 0; t < TT; t++) {
        // ---- GAT layer 1 (2 heads x 32) ----
        k_gat1_node<<<gN, B, 0, stream>>>(x, W1, as1, ad1, h1, a1s, a1d, m1, den1, agg1, t);
        k_e1_max<<<gE, B, 0, stream>>>(ei, a1s, a1d, m1, t);
        k_e1_sum<<<gE, B, 0, stream>>>(ei, a1s, a1d, m1, den1, t);
        k_e1_msg<<<gE64, B, 0, stream>>>(ei, a1s, a1d, m1, den1, h1, agg1, t);
        k_fin1<<<gN64, B, 0, stream>>>(agg1, b1, x2h);
        // ---- GAT layer 2 linear (WMMA) + attention (1 head x 32) ----
        k_gat2_gemm<<<gTile, 32, 0, stream>>>(x2h, W2T, h2);
        k_gat2_node<<<gN, B, 0, stream>>>(h2, as2, ad2, a2s, a2d, m2, den2, agg2);
        k_e2_max<<<gE, B, 0, stream>>>(ei, a2s, a2d, m2, t);
        k_e2_sum<<<gE, B, 0, stream>>>(ei, a2s, a2d, m2, den2, t);
        k_e2_msg<<<gE32, B, 0, stream>>>(ei, a2s, a2d, m2, den2, h2, agg2, t);
        k_fin2<<<gN32, B, 0, stream>>>(agg2, b2, ench);
        // ---- fused GRU step (WMMA) ----
        k_gru<<<gTile, 32, 0, stream>>>(ench, hst, hsth, wihh, whhh, bih, bhh);
    }
    k_fc<<<gN, B, 0, stream>>>(hst, fw, fb, out);
}
